// EgoMotionHead_64424509440414
// MI455X (gfx1250) — matrix operand order repaired and checked by reading, not compiled, not run on hardware
//
#include <hip/hip_runtime.h>
#include <math.h>

// ---------------------------------------------------------------------------
// EgoMotionHead on MI455X (gfx1250).  B=32, N=1024, C=128.
// Sinkhorn-with-slack collapses to potentials u,v:  la = A + u_i + v_j.
// Affinity A materialized once (128 MB, L2-resident) via f16 WMMA, then
// streamed (vectorized b128) for 10 logsumexp passes + 1 final perm pass.
// Per-batch weighted Procrustes via 3x3 Jacobi SVD.
// ---------------------------------------------------------------------------

typedef __attribute__((ext_vector_type(16))) _Float16 v16h;
typedef __attribute__((ext_vector_type(8)))  _Float16 v8h;
typedef __attribute__((ext_vector_type(8)))  float    v8f;

#define B_ 32
#define N_ 1024
#define C_ 128

// -------------------------- helpers ----------------------------------------

__device__ __forceinline__ float wave_max(float x) {
  #pragma unroll
  for (int off = 16; off > 0; off >>= 1) x = fmaxf(x, __shfl_xor(x, off, 32));
  return x;
}
__device__ __forceinline__ float wave_sum(float x) {
  #pragma unroll
  for (int off = 16; off > 0; off >>= 1) x += __shfl_xor(x, off, 32);
  return x;
}

// -------------------------- kernels ----------------------------------------

__global__ void cvt_f16_kernel(const float* __restrict__ in,
                               _Float16* __restrict__ out, int n) {
  int i = blockIdx.x * blockDim.x + threadIdx.x;
  int stride = gridDim.x * blockDim.x;
  for (; i < n; i += stride) out[i] = (_Float16)in[i];
}

__global__ void zero_kernel(float* __restrict__ p, int n) {
  int i = blockIdx.x * blockDim.x + threadIdx.x;
  if (i < n) p[i] = 0.0f;
}

// Affinity GEMM: aff[b][i][j] = (softplus(alpha) - max(0, 2 - 2*fs_i.ft_j)) / (exp(beta)+0.02)
// grid = (B, N/16, N/512); block = 256 (8 waves); each wave: 16 rows x 64 cols.
__global__ __launch_bounds__(256)
void affinity_wmma_kernel(const _Float16* __restrict__ fs16,
                          const _Float16* __restrict__ ft16,
                          const float* __restrict__ alpha,
                          const float* __restrict__ beta,
                          float* __restrict__ aff) {
  const int b    = blockIdx.x;
  const int i0   = blockIdx.y * 16;
  const int lane = threadIdx.x & 31;
  const int wave = threadIdx.x >> 5;
  const int j0   = blockIdx.z * 512 + wave * 64;

  const int rsel = lane & 15;          // row (A) / col (B) within the 16-tile
  const int hi   = lane >> 4;          // lane-half select

  v8f acc[4];
  #pragma unroll
  for (int t = 0; t < 4; ++t)
    #pragma unroll
    for (int e = 0; e < 8; ++e) acc[t][e] = 0.0f;

  const _Float16* arow = fs16 + ((size_t)(b * N_ + i0 + rsel)) * C_;
  const int ak0 = hi * 16;             // A: lanes 0-15 K=0..15; lanes 16-31 K=16..31
  const int bk0 = hi * 16;             // B: lanes 0-15 K=0..15; lanes 16-31 K=16..31

  #pragma unroll
  for (int kc = 0; kc < 4; ++kc) {
    const int kbase = kc * 32;
    v8h alo = *(const v8h*)(arow + kbase + ak0);
    v8h ahi = *(const v8h*)(arow + kbase + ak0 + 8);
    v16h afrag;
    #pragma unroll
    for (int e = 0; e < 8; ++e) { afrag[e] = alo[e]; afrag[8 + e] = ahi[e]; }

    #pragma unroll
    for (int t = 0; t < 4; ++t) {
      const _Float16* brow =
          ft16 + ((size_t)(b * N_ + j0 + t * 16 + rsel)) * C_ + kbase + bk0;
      v16h bfrag = *(const v16h*)brow;
      acc[t] = __builtin_amdgcn_wmma_f32_16x16x32_f16(
          false, afrag, false, bfrag, (short)0, acc[t], false, false);
    }
  }

  const float a0 = alpha[0], b0 = beta[0];
  const float sp = fmaxf(a0, 0.0f) + log1pf(expf(-fabsf(a0)));   // softplus
  const float invden = 1.0f / (expf(b0) + 0.02f);

  const int mbase = hi * 8;            // D layout: lanes 0-15 -> M=vgpr, 16-31 -> M=8+vgpr
  const int col   = lane & 15;
  #pragma unroll
  for (int t = 0; t < 4; ++t) {
    #pragma unroll
    for (int r = 0; r < 8; ++r) {
      const int m = mbase + r;
      const float dot = acc[t][r];
      const float fd  = fmaxf(2.0f - 2.0f * dot, 0.0f);
      aff[(size_t)b * N_ * N_ + (size_t)(i0 + m) * N_ + (j0 + t * 16 + col)] =
          (sp - fd) * invden;
    }
  }
}

// Row pass: u[b][i] = -logsumexp_j( aff[b][i][j] + v[b][j] , 0 ). One wave per row.
// Vectorized: each lane handles 4 consecutive columns per step (b128 loads).
__global__ __launch_bounds__(256)
void row_pass_kernel(const float* __restrict__ aff, const float* __restrict__ v,
                     float* __restrict__ u) {
  const int gw   = blockIdx.x * 8 + (threadIdx.x >> 5);
  const int lane = threadIdx.x & 31;
  const int b = gw >> 10;
  const int i = gw & (N_ - 1);
  const float4* row4 = (const float4*)(aff + (size_t)b * N_ * N_ + (size_t)i * N_);
  const float4* v4   = (const float4*)(v + b * N_);

  float m = -INFINITY;
  for (int j = lane; j < N_ / 4; j += 32) {
    const float4 a = row4[j], vv = v4[j];
    m = fmaxf(m, fmaxf(fmaxf(a.x + vv.x, a.y + vv.y),
                       fmaxf(a.z + vv.z, a.w + vv.w)));
  }
  m = wave_max(m);
  m = fmaxf(m, 0.0f);                      // slack column entry = 0
  float s = 0.0f;
  for (int j = lane; j < N_ / 4; j += 32) {
    const float4 a = row4[j], vv = v4[j];
    s += expf(a.x + vv.x - m) + expf(a.y + vv.y - m) +
         expf(a.z + vv.z - m) + expf(a.w + vv.w - m);
  }
  s = wave_sum(s);
  if (lane == 0) u[b * N_ + i] = -(m + logf(s + expf(-m)));
}

// Col pass (partial): branch-free online logsumexp over a 128-row chunk.
// grid = (B, 8); block = 256; each thread owns 4 adjacent columns (b128/row).
__global__ __launch_bounds__(256)
void col_pass_partial_kernel(const float* __restrict__ aff,
                             const float* __restrict__ u,
                             float* __restrict__ pm, float* __restrict__ ps) {
  const int b = blockIdx.x;
  const int chunk = blockIdx.y;
  const int j0 = threadIdx.x * 4;
  const int i0 = chunk * 128;
  const float* ub = u + b * N_;
  const float4* base =
      (const float4*)(aff + (size_t)b * N_ * N_ + (size_t)i0 * N_ + j0);

  float m[4] = {-INFINITY, -INFINITY, -INFINITY, -INFINITY};
  float s[4] = {0.0f, 0.0f, 0.0f, 0.0f};
  for (int r = 0; r < 128; ++r) {
    const float4 a = base[(size_t)r * (N_ / 4)];
    const float ui = ub[i0 + r];
    const float x[4] = {a.x + ui, a.y + ui, a.z + ui, a.w + ui};
    #pragma unroll
    for (int k = 0; k < 4; ++k) {
      const float mn = fmaxf(m[k], x[k]);                // branch-free online LSE
      s[k] = s[k] * expf(m[k] - mn) + expf(x[k] - mn);
      m[k] = mn;
    }
  }
  #pragma unroll
  for (int k = 0; k < 4; ++k) {
    const size_t idx = ((size_t)b * N_ + j0 + k) * 8 + chunk;
    pm[idx] = m[k];
    ps[idx] = s[k];
  }
}

// Combine partials (+ slack row entry = 0): v = -(M + log(sum)).
__global__ void col_combine_kernel(const float* __restrict__ pm,
                                   const float* __restrict__ ps,
                                   float* __restrict__ v, int total) {
  const int idx = blockIdx.x * blockDim.x + threadIdx.x;
  if (idx >= total) return;
  const float* pmv = pm + (size_t)idx * 8;
  const float* psv = ps + (size_t)idx * 8;
  float M = 0.0f;                          // slack row entry
  #pragma unroll
  for (int k = 0; k < 8; ++k) M = fmaxf(M, pmv[k]);
  float S = expf(-M);
  #pragma unroll
  for (int k = 0; k < 8; ++k) S += psv[k] * expf(pmv[k] - M);
  v[idx] = -(M + logf(S));
}

// Final per-row pass: perm = exp(aff + u + v) * (||cs_i - ct_j||^2 < 9).
// Emits row_sum and unnormalized sum_j perm*ct_j. One wave per row, b128 loads.
__global__ __launch_bounds__(256)
void perm_row_kernel(const float* __restrict__ aff, const float* __restrict__ u,
                     const float* __restrict__ v,
                     const float* __restrict__ coor_s,
                     const float* __restrict__ coor_t,
                     float* __restrict__ rs, float* __restrict__ wt) {
  const int gw   = blockIdx.x * 8 + (threadIdx.x >> 5);
  const int lane = threadIdx.x & 31;
  const int b = gw >> 10;
  const int i = gw & (N_ - 1);
  const float4* row4 = (const float4*)(aff + (size_t)b * N_ * N_ + (size_t)i * N_);
  const float4* v4   = (const float4*)(v + b * N_);
  const float  ui  = u[b * N_ + i];
  const float* cs  = coor_s + ((size_t)b * N_ + i) * 3;
  const float sx = cs[0], sy = cs[1], sz = cs[2];
  const float* ctb = coor_t + (size_t)b * N_ * 3;

  float r = 0.0f, ax = 0.0f, ay = 0.0f, az = 0.0f;
  for (int jv = lane; jv < N_ / 4; jv += 32) {
    const float4 a = row4[jv], vv = v4[jv];
    const float la[4] = {a.x + vv.x, a.y + vv.y, a.z + vv.z, a.w + vv.w};
    #pragma unroll
    for (int k = 0; k < 4; ++k) {
      const int j = jv * 4 + k;
      const float tx = ctb[j * 3], ty = ctb[j * 3 + 1], tz = ctb[j * 3 + 2];
      const float dx = sx - tx, dy = sy - ty, dz = sz - tz;
      const float sup = (dx * dx + dy * dy + dz * dz < 9.0f) ? 1.0f : 0.0f;
      const float p = sup * expf(la[k] + ui);
      r += p; ax += p * tx; ay += p * ty; az += p * tz;
    }
  }
  r = wave_sum(r); ax = wave_sum(ax); ay = wave_sum(ay); az = wave_sum(az);
  if (lane == 0) {
    rs[b * N_ + i] = r;
    float* w = wt + ((size_t)b * N_ + i) * 3;
    w[0] = ax; w[1] = ay; w[2] = az;
  }
}

// ---------------- per-batch Procrustes (3x3 SVD) ----------------------------

__device__ __forceinline__ float det3(const float R[3][3]) {
  return R[0][0] * (R[1][1] * R[2][2] - R[1][2] * R[2][1])
       - R[0][1] * (R[1][0] * R[2][2] - R[1][2] * R[2][0])
       + R[0][2] * (R[1][0] * R[2][1] - R[1][1] * R[2][0]);
}

__device__ void procrustes_solve(const float A[3][3], const float csc[3],
                                 const float ctc[3], float* __restrict__ o) {
  // M = A^T A, Jacobi eigendecomposition -> V, eig.
  float M[3][3], V[3][3] = {{1, 0, 0}, {0, 1, 0}, {0, 0, 1}};
  for (int i = 0; i < 3; ++i)
    for (int j = 0; j < 3; ++j) {
      float acc = 0.0f;
      for (int k = 0; k < 3; ++k) acc += A[k][i] * A[k][j];
      M[i][j] = acc;
    }
  const int PP[3] = {0, 0, 1}, QQ[3] = {1, 2, 2};
  for (int sweep = 0; sweep < 10; ++sweep) {
    for (int r = 0; r < 3; ++r) {
      const int p = PP[r], q = QQ[r];
      const float apq = M[p][q];
      if (fabsf(apq) < 1e-20f) continue;
      const float tau = (M[q][q] - M[p][p]) / (2.0f * apq);
      const float t = (tau >= 0.0f ? 1.0f : -1.0f) /
                      (fabsf(tau) + sqrtf(1.0f + tau * tau));
      const float c = 1.0f / sqrtf(1.0f + t * t);
      const float s = t * c;
      for (int k = 0; k < 3; ++k) {       // M <- M J (columns)
        const float mkp = M[k][p], mkq = M[k][q];
        M[k][p] = c * mkp - s * mkq;
        M[k][q] = s * mkp + c * mkq;
      }
      for (int k = 0; k < 3; ++k) {       // M <- J^T M (rows)
        const float mpk = M[p][k], mqk = M[q][k];
        M[p][k] = c * mpk - s * mqk;
        M[q][k] = s * mpk + c * mqk;
      }
      for (int k = 0; k < 3; ++k) {       // V <- V J
        const float vkp = V[k][p], vkq = V[k][q];
        V[k][p] = c * vkp - s * vkq;
        V[k][q] = s * vkp + c * vkq;
      }
    }
  }
  float eig[3] = {M[0][0], M[1][1], M[2][2]};
  // sort eigenpairs descending (SVD order matters for the det-sign flip)
  for (int a = 0; a < 2; ++a)
    for (int bb = a + 1; bb < 3; ++bb)
      if (eig[bb] > eig[a]) {
        const float te = eig[a]; eig[a] = eig[bb]; eig[bb] = te;
        for (int k = 0; k < 3; ++k) {
          const float tv = V[k][a]; V[k][a] = V[k][bb]; V[k][bb] = tv;
        }
      }
  // U columns: u_k = A v_k / s_k
  float U[3][3], sv[3];
  for (int k = 0; k < 3; ++k) {
    sv[k] = sqrtf(fmaxf(eig[k], 0.0f));
    float av[3];
    for (int i = 0; i < 3; ++i)
      av[i] = A[i][0] * V[0][k] + A[i][1] * V[1][k] + A[i][2] * V[2][k];
    const float inv = (sv[k] > 1e-12f) ? 1.0f / sv[k] : 0.0f;
    for (int i = 0; i < 3; ++i) U[i][k] = av[i] * inv;
  }
  if (sv[2] <= 1e-6f * fmaxf(sv[0], 1e-30f)) {  // degenerate: u2 = u0 x u1
    float cx = U[1][0] * U[2][1] - U[2][0] * U[1][1];
    float cy = U[2][0] * U[0][1] - U[0][0] * U[2][1];
    float cz = U[0][0] * U[1][1] - U[1][0] * U[0][1];
    const float n = sqrtf(cx * cx + cy * cy + cz * cz);
    const float inv = (n > 1e-20f) ? 1.0f / n : 0.0f;
    U[0][2] = cx * inv; U[1][2] = cy * inv; U[2][2] = cz * inv;
  }
  // rot = V U^T (flip 3rd column of V if det <= 0)
  float R[3][3];
  for (int i = 0; i < 3; ++i)
    for (int j = 0; j < 3; ++j)
      R[i][j] = V[i][0] * U[j][0] + V[i][1] * U[j][1] + V[i][2] * U[j][2];
  if (det3(R) <= 0.0f) {
    for (int i = 0; i < 3; ++i)
      for (int j = 0; j < 3; ++j)
        R[i][j] = V[i][0] * U[j][0] + V[i][1] * U[j][1] - V[i][2] * U[j][2];
  }
  for (int i = 0; i < 3; ++i) {
    o[i * 4 + 0] = R[i][0];
    o[i * 4 + 1] = R[i][1];
    o[i * 4 + 2] = R[i][2];
    o[i * 4 + 3] = ctc[i] - (R[i][0] * csc[0] + R[i][1] * csc[1] + R[i][2] * csc[2]);
  }
}

__global__ __launch_bounds__(256)
void batch_solve_kernel(const float* __restrict__ rs, const float* __restrict__ wt,
                        const float* __restrict__ coor_s, float* __restrict__ out) {
  const int b = blockIdx.x, tid = threadIdx.x;
  __shared__ float red[256];
  auto reduce = [&](float val) -> float {
    red[tid] = val; __syncthreads();
    for (int s = 128; s > 0; s >>= 1) {
      if (tid < s) red[tid] += red[tid + s];
      __syncthreads();
    }
    const float r = red[0]; __syncthreads();
    return r;
  };

  float wsum = 0, cx = 0, cy = 0, cz = 0, tx = 0, ty = 0, tz = 0;
  for (int i = tid; i < N_; i += 256) {
    const float w = rs[b * N_ + i];
    const float* cs = coor_s + ((size_t)b * N_ + i) * 3;
    const float* a  = wt + ((size_t)b * N_ + i) * 3;
    const float inv = 1.0f / (w + 1e-5f);           // weighted_t = wt/(rs+eps)
    wsum += w;
    cx += cs[0] * w; cy += cs[1] * w; cz += cs[2] * w;
    tx += a[0] * inv * w; ty += a[1] * inv * w; tz += a[2] * inv * w;
  }
  wsum = reduce(wsum);
  cx = reduce(cx); cy = reduce(cy); cz = reduce(cz);
  tx = reduce(tx); ty = reduce(ty); tz = reduce(tz);
  const float invW = 1.0f / (wsum + 1e-5f);
  const float csc[3] = {cx * invW, cy * invW, cz * invW};
  const float ctc[3] = {tx * invW, ty * invW, tz * invW};

  float cov[9] = {0, 0, 0, 0, 0, 0, 0, 0, 0};
  for (int i = tid; i < N_; i += 256) {
    const float w  = rs[b * N_ + i] * invW;
    const float* cs = coor_s + ((size_t)b * N_ + i) * 3;
    const float* a  = wt + ((size_t)b * N_ + i) * 3;
    const float inv = 1.0f / (rs[b * N_ + i] + 1e-5f);
    const float s0 = cs[0] - csc[0], s1 = cs[1] - csc[1], s2 = cs[2] - csc[2];
    const float t0 = a[0] * inv - ctc[0], t1 = a[1] * inv - ctc[1],
                t2 = a[2] * inv - ctc[2];
    cov[0] += s0 * t0 * w; cov[1] += s0 * t1 * w; cov[2] += s0 * t2 * w;
    cov[3] += s1 * t0 * w; cov[4] += s1 * t1 * w; cov[5] += s1 * t2 * w;
    cov[6] += s2 * t0 * w; cov[7] += s2 * t1 * w; cov[8] += s2 * t2 * w;
  }
  for (int k = 0; k < 9; ++k) cov[k] = reduce(cov[k]);

  if (tid == 0) {
    const float A[3][3] = {{cov[0], cov[1], cov[2]},
                           {cov[3], cov[4], cov[5]},
                           {cov[6], cov[7], cov[8]}};
    procrustes_solve(A, csc, ctc, out + (size_t)b * 12);
  }
}

// -------------------------- launcher ---------------------------------------

extern "C" void kernel_launch(void* const* d_in, const int* in_sizes, int n_in,
                              void* d_out, int out_size, void* d_ws, size_t ws_size,
                              hipStream_t stream) {
  const float* fs    = (const float*)d_in[0];
  const float* ft    = (const float*)d_in[1];
  const float* cs    = (const float*)d_in[2];
  const float* ct    = (const float*)d_in[3];
  const float* alpha = (const float*)d_in[4];
  const float* beta  = (const float*)d_in[5];
  float* out = (float*)d_out;

  char* ws = (char*)d_ws;
  size_t off = 0;
  auto alloc = [&](size_t bytes) -> void* {
    void* p = ws + off;
    off += (bytes + 255) & ~(size_t)255;
    return p;
  };
  _Float16* fs16 = (_Float16*)alloc((size_t)B_ * N_ * C_ * 2);
  _Float16* ft16 = (_Float16*)alloc((size_t)B_ * N_ * C_ * 2);
  float* aff = (float*)alloc((size_t)B_ * N_ * N_ * 4);   // 128 MB, L2-resident
  float* u   = (float*)alloc((size_t)B_ * N_ * 4);
  float* v   = (float*)alloc((size_t)B_ * N_ * 4);
  float* pm  = (float*)alloc((size_t)B_ * N_ * 8 * 4);
  float* ps  = (float*)alloc((size_t)B_ * N_ * 8 * 4);
  float* rsb = (float*)alloc((size_t)B_ * N_ * 4);
  float* wtb = (float*)alloc((size_t)B_ * N_ * 3 * 4);

  const int nfeat = B_ * N_ * C_;
  cvt_f16_kernel<<<1024, 256, 0, stream>>>(fs, fs16, nfeat);
  cvt_f16_kernel<<<1024, 256, 0, stream>>>(ft, ft16, nfeat);

  affinity_wmma_kernel<<<dim3(B_, N_ / 16, N_ / 512), 256, 0, stream>>>(
      fs16, ft16, alpha, beta, aff);

  zero_kernel<<<(B_ * N_ + 255) / 256, 256, 0, stream>>>(v, B_ * N_);

  for (int it = 0; it < 5; ++it) {
    row_pass_kernel<<<(B_ * N_) / 8, 256, 0, stream>>>(aff, v, u);
    col_pass_partial_kernel<<<dim3(B_, 8), 256, 0, stream>>>(aff, u, pm, ps);
    col_combine_kernel<<<(B_ * N_ + 255) / 256, 256, 0, stream>>>(pm, ps, v, B_ * N_);
  }

  perm_row_kernel<<<(B_ * N_) / 8, 256, 0, stream>>>(aff, u, v, cs, ct, rsb, wtb);
  batch_solve_kernel<<<B_, 256, 0, stream>>>(rsb, wtb, cs, out);
}
